// MultiHeadSelfAttentionEins_77506979824111
// MI455X (gfx1250) — compile-verified, matrix operand order checked by reading, not run on hardware
//
#include <hip/hip_runtime.h>
#include <hip/hip_bf16.h>

// MI455X / gfx1250 multi-head self-attention with relative position bias.
// All GEMMs (QKV proj, Q·K^T, P·V, out proj) run on v_wmma_f32_16x16x32_bf16
// (wave32, f32 accumulate). Softmax is fp32 (flash-attention online form).
// K/V tile staging uses gfx1250 async global->LDS copies (ASYNCcnt path).

typedef __bf16 bf16;
typedef __attribute__((ext_vector_type(16))) __bf16 v16bf;
typedef __attribute__((ext_vector_type(8)))  float  v8f;

#define HEADS 16
#define DH 64
#define TSEQ 1024
#define BATCH 4
#define DMODEL 1024

union FragBF {
    v16bf v;
    bf16  h[16];
    uint4 q[2];
};

// D = A x B + C, bf16 inputs, f32 accum (CDNA5 WMMA, wave32)
static __device__ inline v8f wmma_bf16(v16bf a, v16bf b, v8f c) {
    return __builtin_amdgcn_wmma_f32_16x16x32_bf16(
        /*neg_a=*/false, a, /*neg_b=*/false, b,
        /*c_mod=*/(short)0, c, /*reuse_a=*/false, /*reuse_b=*/false);
}

// gfx1250 async copy: 16B global -> LDS, tracked by ASYNCcnt (no VGPR roundtrip).
// LDS operand = low 32 bits of the generic shared pointer (ISA: flat->LDS
// mapping uses addr[31:0] as the wave-relative LDS byte address).
static __device__ inline void async_ld_b128(void* lds, const void* g) {
    asm volatile("global_load_async_to_lds_b128 %0, %1, off"
                 :: "v"((unsigned)(uintptr_t)lds), "v"(g)
                 : "memory");
}
static __device__ inline void wait_async0() {
    asm volatile("s_wait_asynccnt 0x0" ::: "memory");
}

// Load a 16x32 A-fragment (or the mirrored 32x16 B-fragment whose columns are
// K-major rows in memory). ISA layout (CDNA5 ISA, 16-bit A 16x32):
//   lanes 0-15 : row = lane,    K halves {k0..k0+7} and {k0+16..k0+23}
//   lanes 16-31: row = lane-16, K halves {k0+8..k0+15} and {k0+24..k0+31}
static __device__ inline v16bf load_frag_k(const bf16* __restrict__ base,
                                           int row, int ld, int k0) {
    int lane = threadIdx.x & 31;
    int r  = row + (lane & 15);
    int kk = k0 + ((lane >> 4) << 3);
    const bf16* p = base + (size_t)r * ld + kk;
    FragBF f;
    f.q[0] = *(const uint4*)(p);
    f.q[1] = *(const uint4*)(p + 16);
    return f.v;
}

// B-fragment whose columns are strided in memory (V tile: K = rows, N = cols).
static __device__ inline v16bf load_fragB_strided(const bf16* __restrict__ base,
                                                  int col, int ld) {
    int lane = threadIdx.x & 31;
    int n    = col + (lane & 15);
    int koff = (lane >> 4) << 3;
    FragBF f;
#pragma unroll
    for (int e = 0; e < 16; ++e) {
        int kk = koff + (e < 8 ? e : e + 8);
        f.h[e] = base[(size_t)kk * ld + n];
    }
    return f.v;
}

// ---------------------------------------------------------------- cvt fp32->bf16
__global__ void cvt_f32_bf16(const float* __restrict__ in,
                             bf16* __restrict__ out, int n) {
    int i = (blockIdx.x * blockDim.x + threadIdx.x) * 4;
    if (i + 3 < n) {
        float4 f = *(const float4*)(in + i);
        out[i + 0] = (bf16)f.x;
        out[i + 1] = (bf16)f.y;
        out[i + 2] = (bf16)f.z;
        out[i + 3] = (bf16)f.w;
    }
}

// ------------------------------------------------------------------- QKV proj
// C[4096,3072] = Xb[4096,1024] * Wqkv[3072,1024]^T, scattered into Q/K/V
// stored [b][h][t][dh] bf16. Column n factors as n = d*48 + sel*16 + head.
// Register blocking: wave computes 32x64 of C (2 A-frags x 4 B-frags ->
// 8 WMMA per 12 b128 loads).
__global__ __launch_bounds__(128) void qkv_proj_kernel(
    const bf16* __restrict__ xb,   // [4096,1024]
    const bf16* __restrict__ wb,   // [3072,1024]
    bf16* __restrict__ qo, bf16* __restrict__ ko, bf16* __restrict__ vo) {
    const int wave = threadIdx.x >> 5;
    const int lane = threadIdx.x & 31;
    const int m0   = blockIdx.x * 128 + wave * 32;   // grid.x = 32
    const int n0   = blockIdx.y * 64;                // grid.y = 48

    v8f acc[2][4] = {};
    for (int k0 = 0; k0 < DMODEL; k0 += 32) {
        v16bf a0 = load_frag_k(xb, m0,      DMODEL, k0);
        v16bf a1 = load_frag_k(xb, m0 + 16, DMODEL, k0);
#pragma unroll
        for (int c = 0; c < 4; ++c) {
            v16bf b = load_frag_k(wb, n0 + c * 16, DMODEL, k0);
            acc[0][c] = wmma_bf16(a0, b, acc[0][c]);
            acc[1][c] = wmma_bf16(a1, b, acc[1][c]);
        }
    }
    const int hh = lane & 15;
#pragma unroll
    for (int c = 0; c < 4; ++c) {
        int nc  = n0 + c * 16;
        int d   = nc / 48;
        int sel = (nc % 48) / 16;                    // 0=q 1=k 2=v
        bf16* dst = (sel == 0) ? qo : (sel == 1) ? ko : vo;
#pragma unroll
        for (int ms = 0; ms < 2; ++ms)
#pragma unroll
            for (int vg = 0; vg < 8; ++vg) {
                int m  = m0 + ms * 16 + vg + ((lane >> 4) << 3);
                int b_ = m >> 10, t = m & (TSEQ - 1);
                dst[(((size_t)b_ * HEADS + hh) * TSEQ + t) * DH + d] =
                    (bf16)acc[ms][c][vg];
            }
    }
}

// -------------------------------------------------------------- flash attention
// grid = (T/64, B*H), block = 128 (4 waves). Each wave owns 16 query rows;
// K/V 32x64 tiles staged in LDS via async global->LDS copies, shared by all
// 4 waves.
__global__ __launch_bounds__(128) void flash_attn_kernel(
    const bf16* __restrict__ Q, const bf16* __restrict__ K,
    const bf16* __restrict__ V,
    const float* __restrict__ rel_bias,   // [2047, 16]
    bf16* __restrict__ out) {             // [4096, 1024] = b,t,(h*dh)
    __shared__ __align__(16) bf16 sK[32 * DH];
    __shared__ __align__(16) bf16 sV[32 * DH];
    __shared__ __align__(16) bf16 sP[4][16 * 32];

    const int bh   = blockIdx.y;             // 0..63
    const int b    = bh >> 4, h = bh & 15;
    const int wave = threadIdx.x >> 5;
    const int lane = threadIdx.x & 31;
    const int q0   = blockIdx.x * 64 + wave * 16;

    const bf16* Qb = Q + (size_t)bh * TSEQ * DH;
    const bf16* Kb = K + (size_t)bh * TSEQ * DH;
    const bf16* Vb = V + (size_t)bh * TSEQ * DH;

    const v16bf qf0 = load_frag_k(Qb, q0, DH, 0);
    const v16bf qf1 = load_frag_k(Qb, q0, DH, 32);

    v8f o[4] = {};
    float mrun[8], lrun[8];
#pragma unroll
    for (int r = 0; r < 8; ++r) { mrun[r] = -1e30f; lrun[r] = 0.f; }

    const float scale = 0.125f;              // dh^-0.5 = 1/8
    const int   iRow0 = q0 + ((lane >> 4) << 3);
    const int   jc    = lane & 15;
    const int   tb    = threadIdx.x * 16;    // byte offset, 2KB per pass

    for (int j0 = 0; j0 < TSEQ; j0 += 32) {
        // async stage: 32x64 bf16 K and V tiles -> LDS (4KB each, 2 passes)
        {
            const char* gk = (const char*)(Kb + (size_t)j0 * DH);
            const char* gv = (const char*)(Vb + (size_t)j0 * DH);
            async_ld_b128((char*)sK + tb,        gk + tb);
            async_ld_b128((char*)sK + tb + 2048, gk + tb + 2048);
            async_ld_b128((char*)sV + tb,        gv + tb);
            async_ld_b128((char*)sV + tb + 2048, gv + tb + 2048);
            if (j0 + 32 < TSEQ) {            // prefetch next tiles toward L2
                __builtin_prefetch(gk + 4096 + tb, 0, 1);
                __builtin_prefetch(gv + 4096 + tb, 0, 1);
            }
            wait_async0();
        }
        __syncthreads();

        // S = Q * K^T for two 16-col subtiles (K rows act as B columns)
        v8f s0 = {}, s1 = {};
        s0 = wmma_bf16(qf0, load_frag_k(sK, 0, DH, 0),  s0);
        s0 = wmma_bf16(qf1, load_frag_k(sK, 0, DH, 32), s0);
        s1 = wmma_bf16(qf0, load_frag_k(sK, 16, DH, 0),  s1);
        s1 = wmma_bf16(qf1, load_frag_k(sK, 16, DH, 32), s1);

        // scale + relative position bias: bias[i,j] = rel_bias[j-i+T-1, h]
#pragma unroll
        for (int vg = 0; vg < 8; ++vg) {
            int i    = iRow0 + vg;
            int rel0 = (j0 + jc) - i + (TSEQ - 1);
            s0[vg] = s0[vg] * scale + rel_bias[(size_t)rel0 * HEADS + h];
            s1[vg] = s1[vg] * scale + rel_bias[(size_t)(rel0 + 16) * HEADS + h];
        }

        // online softmax (row = fixed per VGPR slot; reduce across 16 lanes)
        float alpha[8];
#pragma unroll
        for (int vg = 0; vg < 8; ++vg) {
            float mx = fmaxf(s0[vg], s1[vg]);
#pragma unroll
            for (int off = 1; off < 16; off <<= 1)
                mx = fmaxf(mx, __shfl_xor(mx, off, 32));
            float mnew = fmaxf(mrun[vg], mx);
            alpha[vg]  = __expf(mrun[vg] - mnew);
            float p0   = __expf(s0[vg] - mnew);
            float p1   = __expf(s1[vg] - mnew);
            s0[vg] = p0; s1[vg] = p1;
            float rs = p0 + p1;
#pragma unroll
            for (int off = 1; off < 16; off <<= 1)
                rs += __shfl_xor(rs, off, 32);
            lrun[vg] = lrun[vg] * alpha[vg] + rs;
            mrun[vg] = mnew;
        }
#pragma unroll
        for (int c = 0; c < 4; ++c)
#pragma unroll
            for (int vg = 0; vg < 8; ++vg) o[c][vg] *= alpha[vg];

        // P (C layout) -> LDS -> A-fragment layout (per-wave private bounce)
        {
            bf16* p = sP[wave];
#pragma unroll
            for (int vg = 0; vg < 8; ++vg) {
                int r = vg + ((lane >> 4) << 3);
                p[r * 32 + jc]      = (bf16)s0[vg];
                p[r * 32 + 16 + jc] = (bf16)s1[vg];
            }
        }
        __syncthreads();

        v16bf pf;
        {
            int r  = lane & 15;
            int kk = (lane >> 4) << 3;
            const bf16* pp = sP[wave] + r * 32 + kk;
            FragBF f;
            f.q[0] = *(const uint4*)(pp);
            f.q[1] = *(const uint4*)(pp + 16);
            pf = f.v;
        }

        // O += P(16x32) * V(32x64)
#pragma unroll
        for (int c = 0; c < 4; ++c) {
            v16bf vf = load_fragB_strided(sV, c * 16, DH);
            o[c] = wmma_bf16(pf, vf, o[c]);
        }
        __syncthreads();   // protect sK/sV/sP before next stage
    }

    // normalize and store [b][t][h*64+d] bf16 for the output projection
#pragma unroll
    for (int vg = 0; vg < 8; ++vg) {
        float inv = 1.f / lrun[vg];
        int   t   = q0 + vg + ((lane >> 4) << 3);
        size_t rb = ((size_t)b * TSEQ + t) * DMODEL + h * DH;
#pragma unroll
        for (int c = 0; c < 4; ++c)
            out[rb + c * 16 + (lane & 15)] = (bf16)(o[c][vg] * inv);
    }
}

// ------------------------------------------------------------------- out proj
// Y[4096,1024] (f32) = AO[4096,1024] * Wout[1024,1024]^T  (bf16 WMMA),
// 2 A-frags x 4 B-frags register blocking per wave.
__global__ __launch_bounds__(128) void out_proj_kernel(
    const bf16* __restrict__ ab,   // [4096,1024]
    const bf16* __restrict__ wb,   // [1024,1024]
    float* __restrict__ y) {
    const int wave = threadIdx.x >> 5;
    const int lane = threadIdx.x & 31;
    const int m0   = blockIdx.x * 128 + wave * 32;   // grid.x = 32
    const int n0   = blockIdx.y * 64;                // grid.y = 16

    v8f acc[2][4] = {};
    for (int k0 = 0; k0 < DMODEL; k0 += 32) {
        v16bf a0 = load_frag_k(ab, m0,      DMODEL, k0);
        v16bf a1 = load_frag_k(ab, m0 + 16, DMODEL, k0);
#pragma unroll
        for (int c = 0; c < 4; ++c) {
            v16bf b = load_frag_k(wb, n0 + c * 16, DMODEL, k0);
            acc[0][c] = wmma_bf16(a0, b, acc[0][c]);
            acc[1][c] = wmma_bf16(a1, b, acc[1][c]);
        }
    }
#pragma unroll
    for (int ms = 0; ms < 2; ++ms)
#pragma unroll
        for (int c = 0; c < 4; ++c)
#pragma unroll
            for (int vg = 0; vg < 8; ++vg) {
                int m = m0 + ms * 16 + vg + ((lane >> 4) << 3);
                y[(size_t)m * DMODEL + n0 + c * 16 + (lane & 15)] =
                    acc[ms][c][vg];
            }
}

// ------------------------------------------------------------------- launcher
extern "C" void kernel_launch(void* const* d_in, const int* in_sizes, int n_in,
                              void* d_out, int out_size, void* d_ws, size_t ws_size,
                              hipStream_t stream) {
    const float* x        = (const float*)d_in[0];   // [4,1024,1024]
    const float* w_qkv    = (const float*)d_in[1];   // [3072,1024]
    const float* w_out    = (const float*)d_in[2];   // [1024,1024]
    const float* rel_bias = (const float*)d_in[3];   // [2047,16]
    float* out = (float*)d_out;

    char* ws = (char*)d_ws;
    const size_t MB = 1u << 20;
    bf16* xb    = (bf16*)(ws + 0 * MB);   //  8 MB: x bf16
    bf16* wqkvb = (bf16*)(ws + 8 * MB);   //  6 MB: w_qkv bf16
    bf16* woutb = (bf16*)(ws + 14 * MB);  //  2 MB: w_out bf16
    bf16* qb    = (bf16*)(ws + 16 * MB);  //  8 MB: Q [b][h][t][dh]
    bf16* kb    = (bf16*)(ws + 24 * MB);  //  8 MB: K
    bf16* vb    = (bf16*)(ws + 32 * MB);  //  8 MB: V
    bf16* ao    = (bf16*)(ws + 40 * MB);  //  8 MB: attn out [b][t][h*dh]

    cvt_f32_bf16<<<4096, 256, 0, stream>>>(x, xb, BATCH * TSEQ * DMODEL);
    cvt_f32_bf16<<<3072, 256, 0, stream>>>(w_qkv, wqkvb, 3 * DMODEL * DMODEL);
    cvt_f32_bf16<<<1024, 256, 0, stream>>>(w_out, woutb, DMODEL * DMODEL);

    dim3 gQKV(32, 48);                // M/128, N/64
    qkv_proj_kernel<<<gQKV, 128, 0, stream>>>(xb, wqkvb, qb, kb, vb);

    dim3 gFA(TSEQ / 64, BATCH * HEADS);
    flash_attn_kernel<<<gFA, 128, 0, stream>>>(qb, kb, vb, rel_bias, ao);

    dim3 gOUT(32, 16);                // M/128, N/64
    out_proj_kernel<<<gOUT, 128, 0, stream>>>(ao, woutb, out);
}
